// Attention_2997887172820
// MI455X (gfx1250) — matrix operand order repaired and checked by reading, not compiled, not run on hardware
//
#include <hip/hip_runtime.h>

// ---------------------------------------------------------------------------
// CDNA5 (gfx1250) attention: QKV projection + flash attention, bf16 WMMA.
// wave32; V_WMMA_F32_16X16X32_BF16 (D = A(16x32) * B(32x16) + C, f32 accum)
// Wt tile staging uses GLOBAL_LOAD_ASYNC_TO_LDS_B128 (ASYNCcnt) on gfx1250.
// ---------------------------------------------------------------------------

typedef __bf16 v16bf __attribute__((ext_vector_type(16)));
typedef __bf16 v8bf  __attribute__((ext_vector_type(8)));
typedef float  v8f   __attribute__((ext_vector_type(8)));
typedef float  v4f   __attribute__((ext_vector_type(4)));

#define S_LEN 4096
#define HD    1024
#define NROWS 16384   // B*S

#define USE_ASYNC_LDS 1

__device__ __forceinline__ v8f wmma_bf16(v16bf a, v16bf b, v8f c) {
  // (neg_a, A, neg_b, B, c_mod, C, reuse_a, reuse_b)
  return __builtin_amdgcn_wmma_f32_16x16x32_bf16(false, a, false, b,
                                                 (short)0, c, false, false);
}

// Async 16-byte global -> LDS copy (per-lane addresses), tracked by ASYNCcnt.
__device__ __forceinline__ void async_copy_b128(const __bf16* gsrc, __bf16* lds) {
  unsigned lds_addr = (unsigned)(unsigned long long)lds;  // low 32b = LDS offset
  asm volatile("global_load_async_to_lds_b128 %0, %1, off"
               :: "v"(lds_addr), "v"(gsrc) : "memory");
}

__device__ __forceinline__ void wait_asynccnt0() {
#if __has_builtin(__builtin_amdgcn_s_wait_asynccnt)
  __builtin_amdgcn_s_wait_asynccnt(0);
#else
  asm volatile("s_wait_asynccnt 0" ::: "memory");
#endif
}

// A-operand (16x32 bf16) from row-major src[row][k], row stride ld.
// Lane l16 = row; half selects K base 0/8; elements 0..7 -> K base..base+7,
// elements 8..15 -> K base+16..base+23 (ISA 7.12.2, 16-bit A 16x32).
__device__ __forceinline__ v16bf load_A_bf16(const __bf16* src, int ld,
                                             int row, int kBase, int half) {
  const __bf16* p0 = src + (size_t)row * ld + kBase + half * 8;
  v8bf lo = *(const v8bf*)p0;
  v8bf hi = *(const v8bf*)(p0 + 16);
  v16bf a;
#pragma unroll
  for (int i = 0; i < 8; ++i) { a[i] = lo[i]; a[i + 8] = hi[i]; }
  return a;
}

// B-operand (32x16 bf16): element B[k][n]. Source laid out n-major:
// src[n*ld + k]. Lane l16 = n; half selects K base 0/8; elements 0..7 ->
// K base..base+7, elements 8..15 -> K base+16..base+23 (matches A).
__device__ __forceinline__ v16bf load_B_bf16(const __bf16* src, int ld,
                                             int n, int kBase, int half) {
  const __bf16* p = src + (size_t)n * ld + kBase + half * 8;
  v8bf lo = *(const v8bf*)p;
  v8bf hi = *(const v8bf*)(p + 16);
  v16bf b;
#pragma unroll
  for (int i = 0; i < 8; ++i) { b[i] = lo[i]; b[i + 8] = hi[i]; }
  return b;
}

// ---------------------------------------------------------------------------
// Kernel 1: transpose + convert weights: Wt[p][n][k] = bf16(W_p[k][n])
// ---------------------------------------------------------------------------
__global__ void wt_kernel(const float* __restrict__ Wq,
                          const float* __restrict__ Wk,
                          const float* __restrict__ Wv,
                          __bf16* __restrict__ Wt) {
  int idx = blockIdx.x * 256 + threadIdx.x;          // 3 * 1024 * 1024 total
  int p   = idx >> 20;
  int rem = idx & ((1 << 20) - 1);
  int n   = rem & 1023;        // coalesced read along n
  int k   = rem >> 10;
  const float* W = (p == 0) ? Wq : (p == 1) ? Wk : Wv;
  Wt[((size_t)p << 20) + (size_t)n * 1024 + k] = (__bf16)W[(size_t)k * 1024 + n];
}

// ---------------------------------------------------------------------------
// Kernel 2: QKV projection GEMM. C[16384 x 1024] = x * W + b, bf16 out.
// Block: 256 thr = 8 waves (4 M-groups x 2 N-groups), tile 64 rows x 128 cols.
// Q scaled by 1/sqrt(D)=1/32 at store; V written transposed Vt[b][d][s].
// ---------------------------------------------------------------------------
__global__ __launch_bounds__(256) void qkv_kernel(
    const float* __restrict__ x, const __bf16* __restrict__ Wt3,
    const float* __restrict__ bq, const float* __restrict__ bk,
    const float* __restrict__ bv,
    __bf16* __restrict__ Qo, __bf16* __restrict__ Ko, __bf16* __restrict__ Vto) {
  const int p    = blockIdx.z;
  const int row0 = blockIdx.y * 64;
  const int col0 = blockIdx.x * 128;
  const int tid  = threadIdx.x;
  const int lane = tid & 31, wave = tid >> 5;
  const int l16 = lane & 15, half = lane >> 4;
  const int wv_m = wave & 3, wv_n = wave >> 2;

  const __bf16* W   = Wt3 + ((size_t)p << 20);
  const float* bias = (p == 0) ? bq : (p == 1) ? bk : bv;

  __shared__ __align__(16) __bf16 x_lds[64 * 32];
  __shared__ __align__(16) __bf16 w_lds[128 * 32];

  v8f acc[4] = {};

  for (int kk = 0; kk < 1024; kk += 32) {
    // stage Wt tile (128 n x 32 k bf16): pure copy -> async DMA to LDS
    {
      int nl = tid >> 1, seg = (tid & 1) * 16;
      const __bf16* wp = W + (size_t)(col0 + nl) * 1024 + kk + seg;
#if USE_ASYNC_LDS
      async_copy_b128(wp,     &w_lds[nl * 32 + seg]);
      async_copy_b128(wp + 8, &w_lds[nl * 32 + seg + 8]);
#else
      *(v8bf*)&w_lds[nl * 32 + seg]     = *(const v8bf*)wp;
      *(v8bf*)&w_lds[nl * 32 + seg + 8] = *(const v8bf*)(wp + 8);
#endif
    }
    // stage X tile (64x32 f32 -> bf16): needs conversion, manual path
    {
      int r = tid >> 2, cs = (tid & 3) * 8;
      const float* xp = x + (size_t)(row0 + r) * 1024 + kk + cs;
      v4f f0 = *(const v4f*)xp;
      v4f f1 = *(const v4f*)(xp + 4);
      v8bf h;
#pragma unroll
      for (int i = 0; i < 4; ++i) { h[i] = (__bf16)f0[i]; h[4 + i] = (__bf16)f1[i]; }
      *(v8bf*)&x_lds[r * 32 + cs] = h;
    }
#if USE_ASYNC_LDS
    wait_asynccnt0();
#endif
    __syncthreads();

    // preload A and all B operands, then burst the 4 WMMAs
    v16bf a = load_A_bf16(x_lds, 32, wv_m * 16 + l16, 0, half);
    v16bf bm[4];
#pragma unroll
    for (int n = 0; n < 4; ++n)
      bm[n] = load_B_bf16(w_lds, 32, wv_n * 64 + n * 16 + l16, 0, half);
#pragma unroll
    for (int n = 0; n < 4; ++n)
      acc[n] = wmma_bf16(a, bm[n], acc[n]);
    __syncthreads();
  }

  // bias + store (C layout: M = j + 8*half, N = l16)
#pragma unroll
  for (int n = 0; n < 4; ++n) {
    int col = col0 + wv_n * 64 + n * 16 + l16;
    float bn = bias[col];
#pragma unroll
    for (int j = 0; j < 8; ++j) {
      int rowg = row0 + wv_m * 16 + j + 8 * half;
      float v = acc[n][j] + bn;
      if (p == 0) {
        Qo[(size_t)rowg * HD + col] = (__bf16)(v * 0.03125f);  // 1/sqrt(1024)
      } else if (p == 1) {
        Ko[(size_t)rowg * HD + col] = (__bf16)v;
      } else {
        int bb = rowg >> 12, s = rowg & (S_LEN - 1);
        Vto[((size_t)bb * HD + col) * S_LEN + s] = (__bf16)v;
      }
    }
  }
}

// ---------------------------------------------------------------------------
// Kernel 3: flash attention. Block = 256 thr = 8 waves; 32 queries/block.
// wave w: mg = w&1 (query 16-row group), dg = w>>1 (256-dim group).
// Online softmax done by wave 0 (full EXEC; WMMA never predicated).
// ---------------------------------------------------------------------------
__global__ __launch_bounds__(256) void flash_kernel(
    const __bf16* __restrict__ Q, const __bf16* __restrict__ K,
    const __bf16* __restrict__ Vt, float* __restrict__ out) {
  const int b  = blockIdx.y;
  const int q0 = blockIdx.x * 32;
  const int tid = threadIdx.x;
  const int lane = tid & 31, wave = tid >> 5;
  const int l16 = lane & 15, half = lane >> 4;
  const int mg = wave & 1, dg = wave >> 1;
  const int d0 = dg * 256;

  const __bf16* Qb = Q  + (size_t)b * S_LEN * HD;
  const __bf16* Kb = K  + (size_t)b * S_LEN * HD;
  const __bf16* Vb = Vt + (size_t)b * HD * S_LEN;

  __shared__ __align__(16) float  s_red[8][16][32];   // per-(mg,dg) partials
  __shared__ __align__(16) __bf16 p_lds[2][16][32];   // P tiles (bf16)
  __shared__ float corr_lds[2][16];
  __shared__ float l_lds[2][16];

  // Hoist Q A-operands: 16 rows x 256 features per wave (8 x v16bf).
  v16bf aq[8];
#pragma unroll
  for (int f = 0; f < 8; ++f)
    aq[f] = load_A_bf16(Qb, HD, q0 + mg * 16 + l16, d0 + f * 32, half);

  v8f o_acc[16] = {};
  float m_i = -__builtin_inff();   // live only in wave 0
  float l_i = 0.0f;

  for (int kb = 0; kb < S_LEN; kb += 32) {
    // ---- scores: S_partial[16q x 32k] over this wave's 256 features ----
    v8f s0 = {}, s1 = {};
#pragma unroll
    for (int f = 0; f < 8; ++f) {
      int kB = d0 + f * 32;
      v16bf b0 = load_B_bf16(Kb, HD, kb + l16,      kB, half);
      v16bf b1 = load_B_bf16(Kb, HD, kb + 16 + l16, kB, half);
      s0 = wmma_bf16(aq[f], b0, s0);
      s1 = wmma_bf16(aq[f], b1, s1);
    }
    if (kb + 32 < S_LEN)  // hint next K block toward the caches
      __builtin_prefetch(Kb + (size_t)(kb + 32 + l16) * HD + d0, 0, 0);

    const int pg = mg * 4 + dg;
#pragma unroll
    for (int j = 0; j < 8; ++j) {
      int r = j + 8 * half;
      s_red[pg][r][l16]      = s0[j];
      s_red[pg][r][16 + l16] = s1[j];
    }
    __syncthreads();

    // ---- online softmax: wave 0, one row per lane (mg = half, r = l16) ----
    if (wave == 0) {
      int smg = half, r = l16;
      float mnew = m_i;
#pragma unroll
      for (int c = 0; c < 32; ++c) {
        float v = s_red[smg * 4 + 0][r][c] + s_red[smg * 4 + 1][r][c] +
                  s_red[smg * 4 + 2][r][c] + s_red[smg * 4 + 3][r][c];
        mnew = fmaxf(mnew, v);
      }
      float cf = __expf(m_i - mnew);
      float lsum = 0.0f;
#pragma unroll
      for (int c = 0; c < 32; ++c) {
        float v = s_red[smg * 4 + 0][r][c] + s_red[smg * 4 + 1][r][c] +
                  s_red[smg * 4 + 2][r][c] + s_red[smg * 4 + 3][r][c];
        float e = __expf(v - mnew);
        lsum += e;
        p_lds[smg][r][c] = (__bf16)e;
      }
      l_i = l_i * cf + lsum;
      m_i = mnew;
      corr_lds[smg][r] = cf;
    }
    __syncthreads();

    // ---- rescale O, then O += P * V ----
    v16bf ap = load_A_bf16(&p_lds[mg][0][0], 32, l16, 0, half);
    float cj[8];
#pragma unroll
    for (int j = 0; j < 8; ++j) cj[j] = corr_lds[mg][j + 8 * half];
#pragma unroll
    for (int n = 0; n < 16; ++n)
#pragma unroll
      for (int j = 0; j < 8; ++j) o_acc[n][j] *= cj[j];
#pragma unroll
    for (int n = 0; n < 16; ++n) {
      v16bf bv = load_B_bf16(Vb, S_LEN, d0 + n * 16 + l16, kb, half);
      o_acc[n] = wmma_bf16(ap, bv, o_acc[n]);
    }
    __syncthreads();
  }

  // ---- finalize: O /= l ----
  if (wave == 0) l_lds[half][l16] = l_i;
  __syncthreads();
  float inv[8];
#pragma unroll
  for (int j = 0; j < 8; ++j) inv[j] = 1.0f / l_lds[mg][j + 8 * half];
#pragma unroll
  for (int n = 0; n < 16; ++n) {
    int col = d0 + n * 16 + l16;
#pragma unroll
    for (int j = 0; j < 8; ++j) {
      int rowg = q0 + mg * 16 + j + 8 * half;
      out[((size_t)b * S_LEN + rowg) * HD + col] = o_acc[n][j] * inv[j];
    }
  }
}

// ---------------------------------------------------------------------------
// Workspace layout (bytes): Wt bf16 3*1M elems (6 MB) | Q bf16 32 MB |
// K bf16 32 MB | Vt bf16 32 MB  => ~102 MB total.
// ---------------------------------------------------------------------------
extern "C" void kernel_launch(void* const* d_in, const int* in_sizes, int n_in,
                              void* d_out, int out_size, void* d_ws, size_t ws_size,
                              hipStream_t stream) {
  const float* x  = (const float*)d_in[0];
  const float* Wq = (const float*)d_in[1];
  const float* bq = (const float*)d_in[2];
  const float* Wk = (const float*)d_in[3];
  const float* bk = (const float*)d_in[4];
  const float* Wv = (const float*)d_in[5];
  const float* bv = (const float*)d_in[6];

  __bf16* wt = (__bf16*)d_ws;
  __bf16* Qb = wt + (size_t)3 * 1024 * 1024;
  __bf16* Kb = Qb + (size_t)NROWS * HD;
  __bf16* Vb = Kb + (size_t)NROWS * HD;

  wt_kernel<<<(3 * 1024 * 1024) / 256, 256, 0, stream>>>(Wq, Wk, Wv, wt);

  dim3 g2(1024 / 128, NROWS / 64, 3);
  qkv_kernel<<<g2, 256, 0, stream>>>(x, wt, bq, bk, bv, Qb, Kb, Vb);

  dim3 g3(S_LEN / 32, 4);
  flash_kernel<<<g3, 256, 0, stream>>>(Qb, Kb, Vb, (float*)d_out);
}
